// Attention_1219770712344
// MI455X (gfx1250) — compile-verified, matrix operand order checked by reading, not run on hardware
//
#include <hip/hip_runtime.h>
#include <hip/hip_bf16.h>

typedef __attribute__((ext_vector_type(16))) __bf16 v16bf;
typedef __attribute__((ext_vector_type(8)))  __bf16 v8bf;
typedef __attribute__((ext_vector_type(8)))  float  v8f;

#define ENC_DIM 2048
#define ATT_DIM 512
#define DEC_DIM 512
#define NB      128
#define NP      196
#define NP_PAD  208   // 13 * 16

// ---------------------------------------------------------------------------
// Kernel 1: W_enc [E=2048][A=512] fp32  ->  WT bf16 [A=512][E=2048] (transposed)
// 32x32 LDS tile transpose so both global read and write are coalesced.
// ---------------------------------------------------------------------------
__global__ __launch_bounds__(256) void k_transpose_convert(
    const float* __restrict__ W, unsigned short* __restrict__ WT_raw) {
  __bf16* __restrict__ WT = reinterpret_cast<__bf16*>(WT_raw);
  __shared__ __bf16 tile[32][33];
  const int a0 = blockIdx.x * 32;   // A dimension
  const int e0 = blockIdx.y * 32;   // E dimension
  const int tx = threadIdx.x & 31;
  const int ty = threadIdx.x >> 5;  // 8 row-groups
#pragma unroll
  for (int r = ty; r < 32; r += 8)
    tile[r][tx] = (__bf16)W[(size_t)(e0 + r) * ATT_DIM + a0 + tx];
  __syncthreads();
#pragma unroll
  for (int r = ty; r < 32; r += 8)
    WT[(size_t)(a0 + r) * ENC_DIM + e0 + tx] = tile[tx][r];
}

// ---------------------------------------------------------------------------
// Kernel 2: c0[b][a] = (dec[b] @ W_dec)[a] + b_dec[a] + b_enc[a]
// One block per batch; decoder row staged in LDS; coalesced W_dec reads.
// ---------------------------------------------------------------------------
__global__ __launch_bounds__(256) void k_compute_c0(
    const float* __restrict__ dec, const float* __restrict__ Wdec,
    const float* __restrict__ bdec, const float* __restrict__ benc,
    float* __restrict__ c0) {
  const int b = blockIdx.x;
  const int t = threadIdx.x;
  __shared__ float sd[DEC_DIM];
  sd[t]       = dec[(size_t)b * DEC_DIM + t];
  sd[t + 256] = dec[(size_t)b * DEC_DIM + t + 256];
  __syncthreads();
  float a0 = 0.f, a1 = 0.f;
#pragma unroll 4
  for (int d = 0; d < DEC_DIM; ++d) {
    const float dv = sd[d];
    a0 = fmaf(dv, Wdec[(size_t)d * ATT_DIM + t],       a0);
    a1 = fmaf(dv, Wdec[(size_t)d * ATT_DIM + 256 + t], a1);
  }
  c0[(size_t)b * ATT_DIM + t]       = a0 + bdec[t]       + benc[t];
  c0[(size_t)b * ATT_DIM + 256 + t] = a1 + bdec[256 + t] + benc[256 + t];
}

// ---------------------------------------------------------------------------
// Kernel 3: fused attention scores.
// grid = (13 pixel tiles, 128 batches), 256 threads = 8 waves.
// Each wave owns 4 N-tiles (wave w -> columns [w*64, w*64+64)).
// K-loop: 64 steps of v_wmma_f32_16x16x32_bf16.
// Epilogue: + c0, relu, * W_full, reduce over A -> att[b, p].
// ---------------------------------------------------------------------------
static __device__ inline v16bf load_bfrag(const __bf16* __restrict__ col, int ka) {
  v8bf lo = *(const v8bf*)(col + ka);        // K = ka .. ka+7      (16 bytes)
  v8bf hi = *(const v8bf*)(col + ka + 16);   // K = ka+16 .. ka+23  (16 bytes)
  v16bf r;
#pragma unroll
  for (int i = 0; i < 8; ++i) { r[i] = lo[i]; r[i + 8] = hi[i]; }
  return r;
}

__global__ __launch_bounds__(256) void k_attn_scores(
    const float* __restrict__ enc,            // [B, P, E] fp32
    const unsigned short* __restrict__ WT_raw,// [A, E] bf16 (transposed W_enc)
    const float* __restrict__ c0,             // [B, A]
    const float* __restrict__ Wfull,          // [A]
    const float* __restrict__ bfull,          // scalar
    float* __restrict__ att) {                // [B, NP_PAD]
  const __bf16* __restrict__ WT = reinterpret_cast<const __bf16*>(WT_raw);
  const int b    = blockIdx.y;
  const int p0   = blockIdx.x * 16;
  const int tid  = threadIdx.x;
  const int wave = tid >> 5;
  const int lane = tid & 31;
  const int half = lane >> 4;   // K-half select (ISA A/B fragment layout)
  const int l15  = lane & 15;

  __shared__ float satt[16];
  if (tid < 16) satt[tid] = 0.f;
  __syncthreads();

  // A operand: this lane's pixel row (lanes 0-15 and 16-31 both cover M=0..15)
  int prow = p0 + l15;
  if (prow > NP - 1) prow = NP - 1;           // padded rows: duplicate last row
  const float* __restrict__ arow = enc + ((size_t)b * NP + prow) * ENC_DIM;

  // B operand: column pointers for this wave's 4 N-tiles
  const int nbase = wave * 64;
  const __bf16* __restrict__ bc0 = WT + (size_t)(nbase +  0 + l15) * ENC_DIM;
  const __bf16* __restrict__ bc1 = WT + (size_t)(nbase + 16 + l15) * ENC_DIM;
  const __bf16* __restrict__ bc2 = WT + (size_t)(nbase + 32 + l15) * ENC_DIM;
  const __bf16* __restrict__ bc3 = WT + (size_t)(nbase + 48 + l15) * ENC_DIM;

  v8f acc0 = {}, acc1 = {}, acc2 = {}, acc3 = {};

  for (int kb = 0; kb < ENC_DIM; kb += 32) {
    const int ka = kb + half * 8;
    // --- A fragment: 16 fp32 -> 16 bf16 (K = ka..ka+7, ka+16..ka+23) ---
    float4 f0 = *(const float4*)(arow + ka);
    float4 f1 = *(const float4*)(arow + ka + 4);
    float4 f2 = *(const float4*)(arow + ka + 16);
    float4 f3 = *(const float4*)(arow + ka + 20);
    v16bf a;
    a[0]  = (__bf16)f0.x; a[1]  = (__bf16)f0.y; a[2]  = (__bf16)f0.z; a[3]  = (__bf16)f0.w;
    a[4]  = (__bf16)f1.x; a[5]  = (__bf16)f1.y; a[6]  = (__bf16)f1.z; a[7]  = (__bf16)f1.w;
    a[8]  = (__bf16)f2.x; a[9]  = (__bf16)f2.y; a[10] = (__bf16)f2.z; a[11] = (__bf16)f2.w;
    a[12] = (__bf16)f3.x; a[13] = (__bf16)f3.y; a[14] = (__bf16)f3.z; a[15] = (__bf16)f3.w;
    // --- B fragments + 4 WMMAs ---
    v16bf b0 = load_bfrag(bc0, ka);
    v16bf b1 = load_bfrag(bc1, ka);
    v16bf b2 = load_bfrag(bc2, ka);
    v16bf b3 = load_bfrag(bc3, ka);
    acc0 = __builtin_amdgcn_wmma_f32_16x16x32_bf16(false, a, false, b0, (short)0, acc0, false, false);
    acc1 = __builtin_amdgcn_wmma_f32_16x16x32_bf16(false, a, false, b1, (short)0, acc1, false, false);
    acc2 = __builtin_amdgcn_wmma_f32_16x16x32_bf16(false, a, false, b2, (short)0, acc2, false, false);
    acc3 = __builtin_amdgcn_wmma_f32_16x16x32_bf16(false, a, false, b3, (short)0, acc3, false, false);
  }

  // --- Epilogue: v = relu(acc + c0[col]) * Wfull[col]; reduce over columns ---
  const float* __restrict__ c0b = c0 + (size_t)b * ATT_DIM;
  float pr[8] = {0.f, 0.f, 0.f, 0.f, 0.f, 0.f, 0.f, 0.f};
#pragma unroll
  for (int t = 0; t < 4; ++t) {
    v8f accv = (t == 0) ? acc0 : (t == 1) ? acc1 : (t == 2) ? acc2 : acc3;
    const int col = nbase + t * 16 + l15;
    const float cc = c0b[col];
    const float wf = Wfull[col];
#pragma unroll
    for (int r = 0; r < 8; ++r) {
      float v = accv[r] + cc;
      v = v > 0.f ? v : 0.f;
      pr[r] = fmaf(v, wf, pr[r]);
    }
  }
  // Cross-lane reduce within each 16-lane half (rows differ between halves).
#pragma unroll
  for (int r = 0; r < 8; ++r) {
#pragma unroll
    for (int off = 1; off < 16; off <<= 1)
      pr[r] += __shfl_xor(pr[r], off, 32);
  }
  if (l15 == 0) {
    // C layout: VGPR r, lanes 0-15 -> row r; lanes 16-31 -> row r+8.
#pragma unroll
    for (int r = 0; r < 8; ++r) atomicAdd(&satt[r + half * 8], pr[r]);
  }
  __syncthreads();
  if (tid < 16) {
    const int p = p0 + tid;
    if (p < NP) att[(size_t)b * NP_PAD + p] = satt[tid] + bfull[0];
  }
}

// ---------------------------------------------------------------------------
// Kernel 4: softmax over 196 pixels per batch -> alpha (into d_out)
// ---------------------------------------------------------------------------
__global__ __launch_bounds__(256) void k_softmax(
    const float* __restrict__ att, float* __restrict__ alpha) {
  const int b = blockIdx.x;
  const int t = threadIdx.x;
  __shared__ float red[256];
  const float v = (t < NP) ? att[(size_t)b * NP_PAD + t] : -3.0e38f;
  red[t] = v;
  __syncthreads();
  for (int s = 128; s > 0; s >>= 1) {
    if (t < s) red[t] = fmaxf(red[t], red[t + s]);
    __syncthreads();
  }
  const float m = red[0];
  __syncthreads();
  const float e = (t < NP) ? __expf(v - m) : 0.f;
  red[t] = e;
  __syncthreads();
  for (int s = 128; s > 0; s >>= 1) {
    if (t < s) red[t] += red[t + s];
    __syncthreads();
  }
  const float inv = 1.0f / red[0];
  if (t < NP) alpha[(size_t)b * NP + t] = e * inv;
}

// ---------------------------------------------------------------------------
// Kernel 5: awe[b,e] = sum_p enc[b,p,e] * alpha[b,p]   (coalesced over e)
// ---------------------------------------------------------------------------
__global__ __launch_bounds__(256) void k_weighted_sum(
    const float* __restrict__ enc, const float* __restrict__ alpha,
    float* __restrict__ awe) {
  const int b = blockIdx.y;
  const int e = blockIdx.x * 256 + threadIdx.x;
  __shared__ float sa[NP];
  if (threadIdx.x < NP) sa[threadIdx.x] = alpha[(size_t)b * NP + threadIdx.x];
  __syncthreads();
  const float* __restrict__ base = enc + (size_t)b * NP * ENC_DIM + e;
  float acc = 0.f;
#pragma unroll 4
  for (int p = 0; p < NP; ++p) acc = fmaf(base[(size_t)p * ENC_DIM], sa[p], acc);
  awe[(size_t)b * ENC_DIM + e] = acc;
}

// ---------------------------------------------------------------------------
extern "C" void kernel_launch(void* const* d_in, const int* in_sizes, int n_in,
                              void* d_out, int out_size, void* d_ws, size_t ws_size,
                              hipStream_t stream) {
  const float* enc    = (const float*)d_in[0];  // [128,196,2048]
  const float* dec    = (const float*)d_in[1];  // [128,512]
  const float* W_enc  = (const float*)d_in[2];  // [2048,512]
  const float* b_enc  = (const float*)d_in[3];  // [512]
  const float* W_dec  = (const float*)d_in[4];  // [512,512]
  const float* b_dec  = (const float*)d_in[5];  // [512]
  const float* W_full = (const float*)d_in[6];  // [512]
  const float* b_full = (const float*)d_in[7];  // scalar

  float* awe   = (float*)d_out;                 // [128*2048]
  float* alpha = (float*)d_out + NB * ENC_DIM;  // [128*196]

  char* ws = (char*)d_ws;
  unsigned short* WT = (unsigned short*)ws;                               // 2 MB bf16
  float* c0  = (float*)(ws + (size_t)ATT_DIM * ENC_DIM * 2);              // 256 KB
  float* att = (float*)(ws + (size_t)ATT_DIM * ENC_DIM * 2 + (size_t)NB * ATT_DIM * 4); // [B, NP_PAD]

  k_transpose_convert<<<dim3(ATT_DIM / 32, ENC_DIM / 32), 256, 0, stream>>>(W_enc, WT);
  k_compute_c0<<<NB, 256, 0, stream>>>(dec, W_dec, b_dec, b_enc, c0);
  k_attn_scores<<<dim3(NP_PAD / 16, NB), 256, 0, stream>>>(enc, WT, c0, W_full, b_full, att);
  k_softmax<<<NB, 256, 0, stream>>>(att, alpha);
  k_weighted_sum<<<dim3(ENC_DIM / 256, NB), 256, 0, stream>>>(enc, alpha, awe);
}
